// DenseMoE_7722351198590
// MI455X (gfx1250) — matrix operand order repaired and checked by reading, not compiled, not run on hardware
//
#include <hip/hip_runtime.h>
#include <hip/hip_bf16.h>

// ---------------- problem constants (match reference) ----------------
#define BB 8192   // batch
#define EE 16     // experts
#define DD 128    // per-expert input dim (= K of expert GEMM)
#define UU 256    // units (= N of expert GEMM)
#define FF 256    // gating feature dim
#define GG 4      // gates
#define KTOP 6    // top-k

typedef __attribute__((ext_vector_type(16))) __bf16 v16bf;
typedef __attribute__((ext_vector_type(8)))  __bf16 v8bf;
typedef __attribute__((ext_vector_type(8)))  float  v8f;
typedef __attribute__((ext_vector_type(8)))  unsigned short v8us;
typedef __attribute__((ext_vector_type(4)))  int    v4i;

// pointer-to-AS1 / AS3 v4i (attribute applies to pointee type)
typedef __attribute__((address_space(1))) v4i* gv4p;
typedef __attribute__((address_space(3))) v4i* lv4p;

// fp32 -> bf16 round-to-nearest-even, returned as 16-bit pattern
__device__ __forceinline__ unsigned bf16_rne_bits(float f) {
  unsigned u = __float_as_uint(f);
  return (u + 0x7FFFu + ((u >> 16) & 1u)) >> 16;
}
// split fp32 into hi/lo bf16 (bf16x3 GEMM trick)
__device__ __forceinline__ void split_bf16(float f, unsigned short& h, unsigned short& l) {
  unsigned hb = bf16_rne_bits(f);
  h = (unsigned short)hb;
  float fh = __uint_as_float(hb << 16);
  l = (unsigned short)bf16_rne_bits(f - fh);
}

__device__ __forceinline__ v16bf cat16(v8bf a, v8bf b) {
  return __builtin_shufflevector(a, b, 0,1,2,3,4,5,6,7,8,9,10,11,12,13,14,15);
}

__device__ __forceinline__ void load16f(const float* __restrict__ p, float4* r) {
  r[0] = *(const float4*)(p + 0);
  r[1] = *(const float4*)(p + 4);
  r[2] = *(const float4*)(p + 8);
  r[3] = *(const float4*)(p + 12);
}

// =====================================================================
// Kernel 1: per-expert GEMM  x[B,E*D] @ W[E,D,U] -> eo[B,U,E]
// tile: 128M x 128N, K=128 in 4 steps of BK=32. 256 thr = 8 wave32.
// LDS: bf16 hi/lo split tiles, row pitch 40 ushorts (80B, 16B aligned).
// Software pipelined: next k-tile's global loads issue before the WMMAs.
// =====================================================================
__global__ __launch_bounds__(256)
void moe_expert_gemm(const float* __restrict__ x,
                     const float* __restrict__ wk,
                     const float* __restrict__ bias,
                     float* __restrict__ eo) {
  __shared__ unsigned short AsH[128][40];
  __shared__ unsigned short AsL[128][40];
  __shared__ unsigned short BsH[128][40];   // stored transposed: [n][k]
  __shared__ unsigned short BsL[128][40];

  const int tid  = threadIdx.x;
  const int lane = tid & 31;
  const int wave = tid >> 5;
  const int half = lane >> 4;
  const int l16  = lane & 15;
  const int m0   = blockIdx.x * 128;
  const int n0   = blockIdx.y * 128;
  const int e    = blockIdx.z;

  v8f acc[8];
  #pragma unroll
  for (int i = 0; i < 8; ++i)
    #pragma unroll
    for (int r = 0; r < 8; ++r) acc[i][r] = 0.0f;

  // staging assignments (per k-step each thread moves 16 f32 of A and B)
  const int am  = tid >> 1;          // A: row within tile
  const int akh = (tid & 1) << 4;    // A: k offset 0/16
  const int bk  = tid >> 3;          // B: k row (0..31)
  const int bub = (tid & 7) << 4;    // B: u offset (0..112)

  const float* agp = x  + (size_t)(m0 + am) * (EE * DD) + e * DD + akh;
  const float* bgp = wk + (size_t)e * (DD * UU) + (size_t)bk * UU + n0 + bub;

  float4 ar[4], br[4];
  load16f(agp, ar);                  // prologue: k-tile 0 -> registers
  load16f(bgp, br);

  #pragma unroll 1
  for (int kt = 0; kt < 4; ++kt) {
    // ---- convert current registers -> split-bf16 LDS tiles ----
    {
      float v[16] = {ar[0].x, ar[0].y, ar[0].z, ar[0].w, ar[1].x, ar[1].y, ar[1].z, ar[1].w,
                     ar[2].x, ar[2].y, ar[2].z, ar[2].w, ar[3].x, ar[3].y, ar[3].z, ar[3].w};
      v8us h0, h1, lo0, lo1;
      #pragma unroll
      for (int i = 0; i < 8; ++i) {
        unsigned short h, l; split_bf16(v[i], h, l); h0[i] = h; lo0[i] = l;
      }
      #pragma unroll
      for (int i = 0; i < 8; ++i) {
        unsigned short h, l; split_bf16(v[8 + i], h, l); h1[i] = h; lo1[i] = l;
      }
      *(v8us*)&AsH[am][akh]     = h0;
      *(v8us*)&AsH[am][akh + 8] = h1;
      *(v8us*)&AsL[am][akh]     = lo0;
      *(v8us*)&AsL[am][akh + 8] = lo1;
    }
    {
      float v[16] = {br[0].x, br[0].y, br[0].z, br[0].w, br[1].x, br[1].y, br[1].z, br[1].w,
                     br[2].x, br[2].y, br[2].z, br[2].w, br[3].x, br[3].y, br[3].z, br[3].w};
      #pragma unroll
      for (int i = 0; i < 16; ++i) {
        unsigned short h, l; split_bf16(v[i], h, l);
        BsH[bub + i][bk] = h;
        BsL[bub + i][bk] = l;
      }
    }
    __syncthreads();

    // ---- issue NEXT k-tile global loads (overlap with WMMAs below) ----
    if (kt < 3) {
      load16f(agp + (kt + 1) * 32, ar);
      load16f(bgp + (size_t)(kt + 1) * 32 * UU, br);
    }

    // ---- A fragments (lane m; elems 0-15 <-> K=16h..16h+15, matching B) ----
    const int mr = wave * 16 + l16;
    v16bf aH = cat16(*(const v8bf*)&AsH[mr][16 * half], *(const v8bf*)&AsH[mr][16 * half + 8]);
    v16bf aL = cat16(*(const v8bf*)&AsL[mr][16 * half], *(const v8bf*)&AsL[mr][16 * half + 8]);

    // ---- 8 N-tiles, bf16x3: hi*hi + hi*lo + lo*hi ----
    #pragma unroll
    for (int nt = 0; nt < 8; ++nt) {
      const int nr = nt * 16 + l16;
      v16bf bH = cat16(*(const v8bf*)&BsH[nr][16 * half], *(const v8bf*)&BsH[nr][16 * half + 8]);
      v16bf bL = cat16(*(const v8bf*)&BsL[nr][16 * half], *(const v8bf*)&BsL[nr][16 * half + 8]);
      acc[nt] = __builtin_amdgcn_wmma_f32_16x16x32_bf16(false, aH, false, bH, (short)0, acc[nt], false, false);
      acc[nt] = __builtin_amdgcn_wmma_f32_16x16x32_bf16(false, aH, false, bL, (short)0, acc[nt], false, false);
      acc[nt] = __builtin_amdgcn_wmma_f32_16x16x32_bf16(false, aL, false, bH, (short)0, acc[nt], false, false);
    }
    if (kt < 3) __syncthreads();   // protect LDS before next staging
  }

  // ---- epilogue: bias + LeakyReLU(0.2), store [B,U,E] layout ----
  #pragma unroll
  for (int nt = 0; nt < 8; ++nt) {
    const int u = n0 + nt * 16 + l16;
    const float bv = bias[e * UU + u];
    #pragma unroll
    for (int r = 0; r < 8; ++r) {
      const int m = m0 + wave * 16 + 8 * half + r;   // C layout: elem r <-> row r+8*half
      float vv = acc[nt][r] + bv;
      vv = vv > 0.0f ? vv : 0.2f * vv;
      eo[(size_t)m * (UU * EE) + (size_t)u * EE + e] = vv;
    }
  }
}

// =====================================================================
// Kernel 2: gating + sharp softmax + top-6 + renorm + weighted combine.
// Block = one gate g, 8 wave32 = 8 batch rows. gk[g] (16KB) staged into
// LDS once per block — via CDNA5 async global->LDS when available.
// =====================================================================
#if __has_builtin(__builtin_amdgcn_global_load_async_to_lds_b128) && \
    __has_builtin(__builtin_amdgcn_s_wait_asynccnt)
#define HAVE_ASYNC_LDS 1
#else
#define HAVE_ASYNC_LDS 0
#endif

#define GK_PITCH 20   // floats per row (80B: 16B-aligned, bank-spread)

__global__ __launch_bounds__(256)
void moe_gate_combine(const float* __restrict__ feat,
                      const float* __restrict__ gk,
                      const float* __restrict__ gb,
                      const float* __restrict__ gw,
                      const float* __restrict__ eo,
                      float* __restrict__ out) {
  __shared__ float gks[FF * GK_PITCH];   // gk[g] as [f][e] rows, pitch 20

  const int tid  = threadIdx.x;
  const int lane = tid & 31;
  const int wave = tid >> 5;
  const int g    = blockIdx.x & 3;
  const int b    = ((blockIdx.x >> 2) << 3) + wave;

  // ---- stage gk[g] (256 rows x 16 floats) into LDS, one row per thread ----
  {
    const float* srow = gk + ((size_t)g * FF + tid) * EE;
    float* drow = &gks[tid * GK_PITCH];
#if HAVE_ASYNC_LDS
    #pragma unroll
    for (int c = 0; c < 4; ++c) {
      __builtin_amdgcn_global_load_async_to_lds_b128(
          (gv4p)(unsigned long long)(srow + 4 * c),
          (lv4p)(unsigned int)(unsigned long long)(drow + 4 * c),
          0, 0);
    }
    __builtin_amdgcn_s_wait_asynccnt(0);
#else
    #pragma unroll
    for (int c = 0; c < 4; ++c)
      *(float4*)(drow + 4 * c) = *(const float4*)(srow + 4 * c);
#endif
  }
  __syncthreads();

  // ---- gating matvec: lanes split F, 16 expert accumulators each ----
  float part[16];
  #pragma unroll
  for (int e = 0; e < 16; ++e) part[e] = 0.0f;
  #pragma unroll
  for (int j = 0; j < 8; ++j) {
    const int f = lane + (j << 5);
    const float xv = feat[(size_t)b * FF + f];
    const float* gr = &gks[f * GK_PITCH];
    float4 c0 = *(const float4*)(gr + 0);
    float4 c1 = *(const float4*)(gr + 4);
    float4 c2 = *(const float4*)(gr + 8);
    float4 c3 = *(const float4*)(gr + 12);
    float gv[16] = {c0.x, c0.y, c0.z, c0.w, c1.x, c1.y, c1.z, c1.w,
                    c2.x, c2.y, c2.z, c2.w, c3.x, c3.y, c3.z, c3.w};
    #pragma unroll
    for (int e = 0; e < 16; ++e) part[e] += xv * gv[e];
  }
  #pragma unroll
  for (int e = 0; e < 16; ++e) {
    #pragma unroll
    for (int off = 16; off >= 1; off >>= 1) part[e] += __shfl_xor(part[e], off);
  }

  // ---- relu(g + bias) * global_weights ----
  float logit[16];
  #pragma unroll
  for (int e = 0; e < 16; ++e) {
    float t = part[e] + gb[g * EE + e];
    t = t > 0.0f ? t : 0.0f;
    logit[e] = t * gw[e];
  }

  // ---- softmax(logit / 0.01) ----
  float mx = logit[0];
  #pragma unroll
  for (int e = 1; e < 16; ++e) mx = fmaxf(mx, logit[e]);
  float p[16]; float s = 0.0f;
  #pragma unroll
  for (int e = 0; e < 16; ++e) { p[e] = __expf((logit[e] - mx) * 100.0f); s += p[e]; }
  const float inv = 1.0f / s;
  #pragma unroll
  for (int e = 0; e < 16; ++e) p[e] *= inv;

  // ---- top-6 via masked argmax (fully unrolled -> cndmask chains) ----
  int idx[KTOP]; float tv[KTOP];
  unsigned used = 0;
  #pragma unroll
  for (int k = 0; k < KTOP; ++k) {
    float best = -1.0f; int bi = 0;
    #pragma unroll
    for (int e = 0; e < 16; ++e) {
      bool ok = !((used >> e) & 1u) && (p[e] > best);
      best = ok ? p[e] : best;
      bi   = ok ? e    : bi;
    }
    idx[k] = bi; tv[k] = best;
    used |= (1u << bi);
  }

  // ---- renormalize: softmax(top_v / 0.01); tv[0] is the max ----
  float w[KTOP]; float s2 = 0.0f;
  #pragma unroll
  for (int k = 0; k < KTOP; ++k) { w[k] = __expf((tv[k] - tv[0]) * 100.0f); s2 += w[k]; }
  const float inv2 = 1.0f / s2;
  #pragma unroll
  for (int k = 0; k < KTOP; ++k) w[k] *= inv2;

  // ---- weighted combine over selected experts ----
  const float* eob = eo + (size_t)b * (UU * EE);
  float* ob = out + (size_t)b * (GG * UU) + g * UU;
  #pragma unroll
  for (int i = 0; i < 8; ++i) {
    const int u = lane + (i << 5);
    const float* r = eob + (size_t)u * EE;
    float a = 0.0f;
    #pragma unroll
    for (int k = 0; k < KTOP; ++k) a += w[k] * r[idx[k]];
    ob[u] = a;
  }
}

// =====================================================================
extern "C" void kernel_launch(void* const* d_in, const int* in_sizes, int n_in,
                              void* d_out, int out_size, void* d_ws, size_t ws_size,
                              hipStream_t stream) {
  const float* inputs = (const float*)d_in[0];   // [B, E*D]
  const float* feat   = (const float*)d_in[1];   // [B, F]
  const float* ek     = (const float*)d_in[2];   // [E, D, U]
  const float* eb     = (const float*)d_in[3];   // [E, U]
  const float* gk     = (const float*)d_in[4];   // [G, F, E]
  const float* gb     = (const float*)d_in[5];   // [G, E]
  const float* gw     = (const float*)d_in[6];   // [E]

  float* out = (float*)d_out;                          // concat [B, G*U]
  float* eo  = out + (size_t)BB * GG * UU;             // expert_outputs [B, U, E]

  dim3 grid1(BB / 128, UU / 128, EE);                  // 64 x 2 x 16
  moe_expert_gemm<<<grid1, 256, 0, stream>>>(inputs, ek, eb, eo);

  const int nblocks2 = (BB / 8) * GG;                  // 4096 blocks: (b-chunk, g)
  moe_gate_combine<<<nblocks2, 256, 0, stream>>>(feat, gk, gb, gw, eo, out);
}
